// gcnmask_67370857005189
// MI455X (gfx1250) — matrix operand order repaired and checked by reading, not compiled, not run on hardware
//
#include <hip/hip_runtime.h>
#include <hip/hip_bf16.h>

typedef __attribute__((ext_vector_type(16))) __bf16 v16bf;
typedef __attribute__((ext_vector_type(8)))  __bf16 v8bf;
typedef __attribute__((ext_vector_type(8)))  float  v8f;
typedef __attribute__((__vector_size__(4 * sizeof(int)))) int i32x4;

#define F    128
#define DEG  16
#define AST1 264   // LDS row stride (bf16 elems) for 256-wide A tile (+8 pad)
#define AST2 136   // LDS row stride for 128-wide A tile (+8 pad)

// CDNA5 async global->LDS path (ASYNCcnt) if the toolchain exposes it.
#if defined(__has_builtin)
#if __has_builtin(__builtin_amdgcn_global_load_async_to_lds_b128) && \
    __has_builtin(__builtin_amdgcn_s_wait_asynccnt)
#define HAVE_ASYNC_LDS 1
#endif
#endif

#define AS1 __attribute__((address_space(1)))
#define AS3 __attribute__((address_space(3)))

__device__ inline void copy16_to_lds(__bf16* __restrict__ dst,
                                     const __bf16* __restrict__ src) {
#if defined(HAVE_ASYNC_LDS)
    __builtin_amdgcn_global_load_async_to_lds_b128(
        (AS1 i32x4*)(void*)src, (AS3 i32x4*)(void*)dst, 0, 0);
#else
    *(v8bf*)dst = *(const v8bf*)src;
#endif
}

__device__ inline void wait_fill() {
#if defined(HAVE_ASYNC_LDS)
    __builtin_amdgcn_s_wait_asynccnt(0);
#endif
}

// ---------------------------------------------------------------------------
// Prep: bf16 conversion + packing B matrices into per-lane WMMA fragment
// order so GEMM waves load fragments with contiguous 32B loads.
// B fragment layout (V_WMMA_F32_16X16X32_BF16, B = 32x16 KxN):
//   lane holds column n = lane&15; lanes 0-15 hold K 0..7,16..23, lanes 16-31 K 8..15,24..31;
//   element e (0..15) of the v16bf  ->  K = 32*s + (e>>3)*16 + (lane>>4)*8 + (e&7).
// ---------------------------------------------------------------------------
__global__ void k_cvt_bf16(const float* __restrict__ src, __bf16* __restrict__ dst, int n) {
    int i = blockIdx.x * blockDim.x + threadIdx.x;
    if (i < n) dst[i] = (__bf16)src[i];
}

// weights_mask [256,128] -> packed [j:8][s:8][lane:32][e:16]
__global__ void k_pack_wm(const float* __restrict__ wm, __bf16* __restrict__ wmP) {
    int t = blockIdx.x * blockDim.x + threadIdx.x;
    if (t >= 8 * 8 * 32 * 16) return;
    int e = t & 15, lane = (t >> 4) & 31, s = (t >> 9) & 7, j = t >> 12;
    int n = j * 16 + (lane & 15);
    int k = s * 32 + ((e >> 3) << 4) + ((lane >> 4) << 3) + (e & 7);
    wmP[t] = (__bf16)wm[k * F + n];
}

// weight [128,128] -> packed [j:8][s:4][lane:32][e:16]
__global__ void k_pack_w(const float* __restrict__ w, __bf16* __restrict__ wP) {
    int t = blockIdx.x * blockDim.x + threadIdx.x;
    if (t >= 8 * 4 * 32 * 16) return;
    int e = t & 15, lane = (t >> 4) & 31, s = (t >> 9) & 3, j = t >> 11;
    int n = j * 16 + (lane & 15);
    int k = s * 32 + ((e >> 3) << 4) + ((lane >> 4) << 3) + (e & 7);
    wP[t] = (__bf16)w[k * F + n];
}

// Load a 16x32 bf16 A fragment from an LDS tile (row-major, stride in elems).
// 16-bit A layout: lanes 0-15 (hi=0): elems 0..7 = K+0..7, 8..15 = K+16..23;
//                  lanes 16-31 (hi=1): elems 0..7 = K+8..15, 8..15 = K+24..31.
__device__ inline v16bf load_afrag(const __bf16* tile, int stride, int lane, int k0) {
    int row = lane & 15;
    int hi  = lane >> 4;
    const __bf16* p = tile + row * stride + k0 + hi * 8;
    v8bf lo  = *(const v8bf*)(p);
    v8bf hi8 = *(const v8bf*)(p + 16);
    v16bf a;
#pragma unroll
    for (int i = 0; i < 8; ++i) { a[i] = lo[i]; a[8 + i] = hi8[i]; }
    return a;
}

__device__ inline float fast_sigmoid(float v) {
    return __builtin_amdgcn_rcpf(1.0f + __expf(-v));   // v_exp + v_rcp
}

// Vectorized fill of one 16x256 A tile (cen||nei), 512 x 16B chunks.
// Wave w handles chunks [32w,32w+32) and [256+32w, ...): lanes 0-15 read a
// contiguous 256B run of the cen row, lanes 16-31 a contiguous 256B nei row.
__device__ inline void fill_tile1(__bf16* __restrict__ A,
                                  const __bf16* __restrict__ xbf,
                                  const int* __restrict__ edge_src, long d) {
    const long eb = d * DEG;
    const int tid = threadIdx.x;
#pragma unroll
    for (int h = 0; h < 2; ++h) {
        int ch = tid + h * 256;        // 0..511
        int r  = ch >> 5;              // edge row 0..15
        int cc = ch & 31;              // 16B chunk within the 512B row
        const __bf16* src;
        if (cc < 16) src = xbf + d * F + cc * 8;                               // cen
        else         src = xbf + (long)edge_src[eb + r] * F + (cc - 16) * 8;   // nei
        copy16_to_lds(A + r * AST1 + cc * 8, src);
    }
}

// ---------------------------------------------------------------------------
// Kernel 1: per-destination-node mask MLP + sigmoid + neighbor aggregation.
// 8 waves per workgroup; wave j owns output columns [16j,16j+16). B fragments
// stay in registers across the node loop. A tile is double-buffered in LDS:
// the gather for node d+stride is issued (async if available) while node d's
// WMMAs run; the ASYNCcnt wait + barrier at loop top is the buffer handoff.
// ---------------------------------------------------------------------------
__global__ __launch_bounds__(256) void k_mask_agg(
    const float*  __restrict__ x,        // [N,128] f32
    const __bf16* __restrict__ xbf,      // [N,128] bf16
    const __bf16* __restrict__ wmP,      // packed weights_mask fragments
    const int*    __restrict__ edge_src, // [N*16]
    float*        __restrict__ xnew,     // [N,128] f32
    __bf16*       __restrict__ xnewbf,   // [N,128] bf16
    int N)
{
    __shared__ __bf16 A[2][16 * AST1];
    const int tid  = threadIdx.x;
    const int wave = tid >> 5;          // 0..7 -> output column tile
    const int lane = tid & 31;
    const int col  = lane & 15;
    const int hi   = lane >> 4;

    // Loop-invariant B fragments for this wave's 16 columns (8 K-steps).
    v16bf bfrag[8];
#pragma unroll
    for (int s = 0; s < 8; ++s)
        bfrag[s] = *(const v16bf*)(wmP + (((wave * 8 + s) * 32 + lane) << 4));

    int buf = 0;
    long d0 = blockIdx.x;
    if (d0 < N) fill_tile1(A[0], xbf, edge_src, d0);

    for (long d = d0; d < N; d += gridDim.x) {
        wait_fill();
        __syncthreads();                         // A[buf] fill complete
        long dn = d + gridDim.x;
        if (dn < N) fill_tile1(A[1 - buf], xbf, edge_src, dn);  // prefetch next

        // Preload all 8 A fragments, then issue WMMAs back-to-back
        // on two independent accumulator chains.
        v16bf afr[8];
#pragma unroll
        for (int s = 0; s < 8; ++s)
            afr[s] = load_afrag(A[buf], AST1, lane, s * 32);

        v8f acc0 = {}, acc1 = {};
#pragma unroll
        for (int s = 0; s < 8; s += 2) {
            acc0 = __builtin_amdgcn_wmma_f32_16x16x32_bf16(
                       false, afr[s],     false, bfrag[s],     (short)0, acc0, false, false);
            acc1 = __builtin_amdgcn_wmma_f32_16x16x32_bf16(
                       false, afr[s + 1], false, bfrag[s + 1], (short)0, acc1, false, false);
        }

        // sigmoid, multiply by nei, reduce over the 16 edges (rows).
        // C layout: elem r -> row hi*8+r, column = col; feature = 16*wave+col.
        float partial = 0.0f;
#pragma unroll
        for (int r = 0; r < 8; ++r) {
            float m  = fast_sigmoid(acc0[r] + acc1[r]);
            int row  = hi * 8 + r;
            float nv = (float)A[buf][row * AST1 + F + wave * 16 + col];
            partial += m * nv;
        }
        partial += __shfl_xor(partial, 16, 32);   // combine row halves

        if (lane < 16) {
            int f   = wave * 16 + col;
            float v = x[d * F + f] + partial;     // x_new = x + agg
            xnew[d * F + f]   = v;
            xnewbf[d * F + f] = (__bf16)v;
        }
        buf ^= 1;
    }
}

// ---------------------------------------------------------------------------
// Kernel 2: support = x_new @ weight  (M tiles of 16 nodes, K=128, N=128).
// ---------------------------------------------------------------------------
__global__ __launch_bounds__(256) void k_support(
    const __bf16* __restrict__ xnewbf,  // [N,128]
    const __bf16* __restrict__ wP,      // packed weight fragments
    float*        __restrict__ support, // [N,128]
    int N, int ntiles)
{
    __shared__ __bf16 A[16 * AST2];
    const int tid  = threadIdx.x;
    const int wave = tid >> 5;
    const int lane = tid & 31;
    const int col  = lane & 15;
    const int hi   = lane >> 4;

    v16bf bfrag[4];
#pragma unroll
    for (int s = 0; s < 4; ++s)
        bfrag[s] = *(const v16bf*)(wP + (((wave * 4 + s) * 32 + lane) << 4));

    for (int t = blockIdx.x; t < ntiles; t += gridDim.x) {
        // fill cols 0..63 of the 16x128 tile: 128 x 16B chunks (16 rows x 8 chunks)
        if (tid < 128) {
            int r  = tid >> 3;
            int cc = tid & 7;
            long row = (long)t * 16 + r;
            if (row >= N) row = N - 1;                // clamp (harmless reread)
            copy16_to_lds(A + r * AST2 + cc * 8, xnewbf + row * F + cc * 8);
        }
        wait_fill();
        __syncthreads();

        v16bf afr[4];
#pragma unroll
        for (int s = 0; s < 4; ++s)
            afr[s] = load_afrag(A, AST2, lane, s * 32);

        v8f acc0 = {}, acc1 = {};
        acc0 = __builtin_amdgcn_wmma_f32_16x16x32_bf16(
                   false, afr[0], false, bfrag[0], (short)0, acc0, false, false);
        acc1 = __builtin_amdgcn_wmma_f32_16x16x32_bf16(
                   false, afr[1], false, bfrag[1], (short)0, acc1, false, false);
        acc0 = __builtin_amdgcn_wmma_f32_16x16x32_bf16(
                   false, afr[2], false, bfrag[2], (short)0, acc0, false, false);
        acc1 = __builtin_amdgcn_wmma_f32_16x16x32_bf16(
                   false, afr[3], false, bfrag[3], (short)0, acc1, false, false);

#pragma unroll
        for (int r = 0; r < 8; ++r) {
            long row = (long)t * 16 + hi * 8 + r;
            if (row < N)
                support[row * F + wave * 16 + col] = acc0[r] + acc1[r];
        }
        __syncthreads();
    }
}

// ---------------------------------------------------------------------------
// Kernel 3: out[d] = bias + sum_e adj[e] * support[src[e]]  (float4 per lane;
// each wave gathers one full 512B row per edge -> coalesced b128 loads).
// ---------------------------------------------------------------------------
__global__ __launch_bounds__(256) void k_out(
    const float* __restrict__ support,
    const float* __restrict__ adj_vals,
    const int*   __restrict__ edge_src,
    const float* __restrict__ bias,
    float*       __restrict__ out, int N)
{
    int d = blockIdx.x * 8 + (threadIdx.x >> 5);  // 8 nodes per block
    int c = threadIdx.x & 31;                     // float4 index (32*4 = 128 feats)
    if (d >= N) return;
    long eb = (long)d * DEG;
    const float4* sup4 = (const float4*)support;
    float4 acc = ((const float4*)bias)[c];
#pragma unroll 4
    for (int e = 0; e < DEG; ++e) {
        int   s = edge_src[eb + e];
        float a = adj_vals[eb + e];
        float4 v = sup4[(long)s * 32 + c];
        acc.x += a * v.x; acc.y += a * v.y; acc.z += a * v.z; acc.w += a * v.w;
    }
    ((float4*)out)[(long)d * 32 + c] = acc;
}

// ---------------------------------------------------------------------------
extern "C" void kernel_launch(void* const* d_in, const int* in_sizes, int n_in,
                              void* d_out, int out_size, void* d_ws, size_t ws_size,
                              hipStream_t stream) {
    const float* x        = (const float*)d_in[0];   // [N,128]
    const float* weight   = (const float*)d_in[1];   // [128,128]
    const float* bias     = (const float*)d_in[2];   // [128]
    const float* wmask    = (const float*)d_in[3];   // [256,128]
    const float* adj_vals = (const float*)d_in[4];   // [E]
    const int*   edge_src = (const int*)d_in[5];     // [E]
    // d_in[6] = edge_dst, implicit (repeat(arange(N),16))

    const int N = in_sizes[0] / F;
    const int ntiles = (N + 15) / 16;

    // workspace carve-up (256B aligned)
    char* base = (char*)d_ws;
    size_t off = 0;
    auto carve = [&](size_t bytes) -> char* {
        char* p = base + off;
        off += (bytes + 255) & ~(size_t)255;
        return p;
    };
    __bf16* xbf     = (__bf16*)carve((size_t)N * F * 2);
    __bf16* wmP     = (__bf16*)carve(8 * 8 * 32 * 16 * 2);
    __bf16* wP      = (__bf16*)carve(8 * 4 * 32 * 16 * 2);
    float*  xnew    = (float*) carve((size_t)N * F * 4);
    __bf16* xnewbf  = (__bf16*)carve((size_t)N * F * 2);
    float*  support = (float*) carve((size_t)N * F * 4);
    (void)ws_size;

    // prep
    {
        int n = N * F;
        k_cvt_bf16<<<(n + 255) / 256, 256, 0, stream>>>(x, xbf, n);
        k_pack_wm<<<(8 * 8 * 32 * 16 + 255) / 256, 256, 0, stream>>>(wmask, wmP);
        k_pack_w <<<(8 * 4 * 32 * 16 + 255) / 256, 256, 0, stream>>>(weight, wP);
    }

    // per-edge mask MLP + aggregation -> x_new
    {
        int blocks = N < 4096 ? N : 4096;
        k_mask_agg<<<blocks, 256, 0, stream>>>(x, xbf, wmP, edge_src, xnew, xnewbf, N);
    }

    // support = x_new @ weight
    {
        int blocks = ntiles < 2048 ? ntiles : 2048;
        k_support<<<blocks, 256, 0, stream>>>(xnewbf, wP, support, N, ntiles);
    }

    // out = segment_sum(adj * support[src]) + bias
    {
        int blocks = (N + 7) / 8;
        k_out<<<blocks, 256, 0, stream>>>(support, adj_vals, edge_src, bias,
                                          (float*)d_out, N);
    }
}
